// SimpleSelfAttention_19078244728943
// MI455X (gfx1250) — matrix operand order repaired and checked by reading, not compiled, not run on hardware
//
#include <hip/hip_runtime.h>
#include <hip/hip_bf16.h>

// ---------------------------------------------------------------------------
// Flash-attention for x[4,4096,512] fp32:  y = softmax(x x^T) x
// bf16 WMMA (v_wmma_f32_16x16x32_bf16) with fp32 accumulation, online softmax.
// ---------------------------------------------------------------------------

typedef __attribute__((ext_vector_type(16))) __bf16        v16bf;
typedef __attribute__((ext_vector_type(8)))  float          v8f;
typedef __attribute__((ext_vector_type(16))) unsigned short v16u;
typedef __attribute__((ext_vector_type(8)))  unsigned short v8u;

constexpr int SEQ = 4096;   // T
constexpr int DIM = 512;    // K / head dim
constexpr int TC  = 128;    // keys per outer tile
constexpr int NW  = 8;      // waves per workgroup

__device__ __forceinline__ unsigned short f2bf(float f) {
    unsigned int u = __builtin_bit_cast(unsigned int, f);
    u += 0x7fffu + ((u >> 16) & 1u);            // round-to-nearest-even
    return (unsigned short)(u >> 16);
}

// A-matrix fragment (16x32 bf16): lane holds row m = lane&15;
// K = (lane>=16 ? 16 : 0) + 0..15 contiguous.  caller passes p = &row[kb + half*16].
__device__ __forceinline__ v16bf load_a_frag(const unsigned short* p) {
    v16u u = *(const v16u*)p;
    return __builtin_bit_cast(v16bf, u);
}

// B-matrix fragment (32x16 bf16): lane holds col n = lane&15;
// K = (lane>=16 ? 16 : 0) + 0..15 contiguous.  caller passes &col[kb + half*16].
__device__ __forceinline__ v16bf load_b_frag(const unsigned short* p) {
    v16u u = *(const v16u*)p;
    return __builtin_bit_cast(v16bf, u);
}

__device__ __forceinline__ v8f wmma_bf16(v16bf a, v16bf b, v8f c) {
    return __builtin_amdgcn_wmma_f32_16x16x32_bf16(false, a, false, b,
                                                   (short)0, c, false, false);
}

// ---------------------------------------------------------------------------
// Pre-pass: x fp32 -> xb bf16 row-major [b][t][d]  and  xt bf16 [b][d][t]
// ---------------------------------------------------------------------------
__global__ __launch_bounds__(256) void cvt_transpose(
        const float* __restrict__ x,
        unsigned short* __restrict__ xb,
        unsigned short* __restrict__ xt) {
    __shared__ unsigned short tile[32][33];
    const int tx = threadIdx.x, ty = threadIdx.y;
    const int d0 = blockIdx.x * 32, t0 = blockIdx.y * 32, b = blockIdx.z;
#pragma unroll
    for (int i = 0; i < 4; ++i) {
        const int t = t0 + ty + i * 8;
        const int d = d0 + tx;
        const float f = x[((size_t)b * SEQ + t) * DIM + d];
        const unsigned short h = f2bf(f);
        xb[((size_t)b * SEQ + t) * DIM + d] = h;
        tile[ty + i * 8][tx] = h;
    }
    __syncthreads();
#pragma unroll
    for (int i = 0; i < 4; ++i) {
        const int d = d0 + ty + i * 8;
        const int t = t0 + tx;
        xt[((size_t)b * DIM + d) * SEQ + t] = tile[tx][ty + i * 8];
    }
}

// ---------------------------------------------------------------------------
// Main flash-attention kernel: one workgroup per (batch, 16-row query block).
// 8 waves; wave w owns key sub-block [s0+16w, s0+16w+16) for QK^T and the
// d-slice [64w, 64w+64) for the PV accumulation.
// ---------------------------------------------------------------------------
__global__ __launch_bounds__(256) void attn_fa(
        const unsigned short* __restrict__ xb,   // [b][t][d] bf16
        const unsigned short* __restrict__ xt,   // [b][d][t] bf16
        float* __restrict__ out) {               // [b][t][d] fp32
    __shared__ alignas(32) unsigned short Plds[16][TC];  // exp'd probs, bf16
    __shared__ float wmax[NW][16];
    __shared__ float wsum[NW][16];
    __shared__ float mrun[16];
    __shared__ float lrun[16];

    const int tid  = threadIdx.x;
    const int w    = tid >> 5;
    const int lane = tid & 31;
    const int half = lane >> 4;
    const int l16  = lane & 15;

    const int batch = blockIdx.x >> 8;          // SEQ/16 = 256 blocks/batch
    const int m0    = (blockIdx.x & 255) << 4;

    const unsigned short* xbB = xb + (size_t)batch * SEQ * DIM;
    const unsigned short* xtB = xt + (size_t)batch * DIM * SEQ;

    if (tid < 16) { mrun[tid] = -__builtin_inff(); lrun[tid] = 0.0f; }

    // Q fragments for all 512 k, kept in registers (16 x 8 VGPRs).
    v16bf qf[16];
    {
        const unsigned short* qrow = xbB + (size_t)(m0 + l16) * DIM + half * 16;
#pragma unroll
        for (int c = 0; c < 16; ++c) qf[c] = load_a_frag(qrow + 32 * c);
    }
    v8f o0 = {}, o1 = {}, o2 = {}, o3 = {};
    __syncthreads();

    for (int s0 = 0; s0 < SEQ; s0 += TC) {
        // ---- S = Q K^T for this wave's 16 keys (full K = 512) ----
        v8f s = {};
        {
            const unsigned short* krow =
                xbB + (size_t)(s0 + w * 16 + l16) * DIM + half * 16;
#pragma unroll
            for (int c = 0; c < 16; ++c)
                s = wmma_bf16(qf[c], load_b_frag(krow + 32 * c), s);
        }
        // ---- per-wave row max -> LDS ----
#pragma unroll
        for (int r = 0; r < 8; ++r) {
            float v = s[r];
            v = fmaxf(v, __shfl_xor(v, 1, 32));
            v = fmaxf(v, __shfl_xor(v, 2, 32));
            v = fmaxf(v, __shfl_xor(v, 4, 32));
            v = fmaxf(v, __shfl_xor(v, 8, 32));
            if (l16 == 0) wmax[w][r + 8 * half] = v;
        }
        __syncthreads();
        // ---- combine maxes, exponentiate, partial sums, rescale O ----
#pragma unroll
        for (int r = 0; r < 8; ++r) {
            const int row = r + 8 * half;
            const float mo = mrun[row];
            float mn = mo;
#pragma unroll
            for (int ww = 0; ww < NW; ++ww) mn = fmaxf(mn, wmax[ww][row]);
            const float alpha = __expf(mo - mn);   // 0 on first tile (mo=-inf)
            const float p = __expf(s[r] - mn);
            Plds[row][w * 16 + l16] = f2bf(p);
            float rs = p;
            rs += __shfl_xor(rs, 1, 32);
            rs += __shfl_xor(rs, 2, 32);
            rs += __shfl_xor(rs, 4, 32);
            rs += __shfl_xor(rs, 8, 32);
            if (l16 == 0) wsum[w][row] = rs;
            o0[r] *= alpha; o1[r] *= alpha; o2[r] *= alpha; o3[r] *= alpha;
        }
        __syncthreads();
        // ---- update running softmax stats (16 threads, overlaps with PV) ----
        if (tid < 16) {
            const float mo = mrun[tid];
            float mn = mo;
#pragma unroll
            for (int ww = 0; ww < NW; ++ww) mn = fmaxf(mn, wmax[ww][tid]);
            float acc = 0.0f;
#pragma unroll
            for (int ww = 0; ww < NW; ++ww) acc += wsum[ww][tid];
            lrun[tid] = lrun[tid] * __expf(mo - mn) + acc;
            mrun[tid] = mn;
        }
        // ---- O += P V for this wave's 64-wide d-slice ----
        {
            const unsigned short* vcol =
                xtB + (size_t)(w * 64 + l16) * SEQ + s0 + half * 16;
#pragma unroll
            for (int c = 0; c < 4; ++c) {
                const v16bf pa = load_a_frag(&Plds[l16][32 * c + half * 16]);
                o0 = wmma_bf16(pa, load_b_frag(vcol + 0 * SEQ + 32 * c), o0);
                o1 = wmma_bf16(pa, load_b_frag(vcol + 16 * SEQ + 32 * c), o1);
                o2 = wmma_bf16(pa, load_b_frag(vcol + 32 * SEQ + 32 * c), o2);
                o3 = wmma_bf16(pa, load_b_frag(vcol + 48 * SEQ + 32 * c), o3);
            }
        }
        __syncthreads();
    }

    // ---- epilogue: divide by l, store fp32 ----
#pragma unroll
    for (int r = 0; r < 8; ++r) {
        const int row = r + 8 * half;
        const float inv = 1.0f / lrun[row];
        float* orow = out + ((size_t)batch * SEQ + (m0 + row)) * DIM
                          + w * 64 + l16;
        orow[0]  = o0[r] * inv;
        orow[16] = o1[r] * inv;
        orow[32] = o2[r] * inv;
        orow[48] = o3[r] * inv;
    }
}

// ---------------------------------------------------------------------------
extern "C" void kernel_launch(void* const* d_in, const int* in_sizes, int n_in,
                              void* d_out, int out_size, void* d_ws, size_t ws_size,
                              hipStream_t stream) {
    (void)in_sizes; (void)n_in; (void)out_size; (void)ws_size;
    const float* x = (const float*)d_in[0];
    float* out = (float*)d_out;

    unsigned short* xb = (unsigned short*)d_ws;                 // 16 MB bf16 [b][t][d]
    unsigned short* xt = xb + (size_t)4 * SEQ * DIM;            // 16 MB bf16 [b][d][t]

    dim3 g1(DIM / 32, SEQ / 32, 4), b1(32, 8);
    cvt_transpose<<<g1, b1, 0, stream>>>(x, xb, xt);

    attn_fa<<<dim3(4 * (SEQ / 16)), dim3(256), 0, stream>>>(xb, xt, out);
}
